// VoronoiSpatialPartitioner_80393197846795
// MI455X (gfx1250) — matrix operand order repaired and checked by reading, not compiled, not run on hardware
//
#include <hip/hip_runtime.h>
#include <stdint.h>

typedef __attribute__((ext_vector_type(16))) _Float16 v16h;
typedef __attribute__((ext_vector_type(8)))  float    v8f;

#define N_PTS     131072
#define B_DIM     128
#define R_DIM     256
#define SIDE      16
#define S_SLICES  128
#define SLICE_PTS (N_PTS / S_SLICES)   // 1024
#define KSTEPS    (SLICE_PTS / 32)     // 32
#define BR        (B_DIM * R_DIM)      // 32768

// ---------------------------------------------------------------------------
// Kernel 0: zero the integer count accumulators (256 entries).
// ---------------------------------------------------------------------------
__global__ void vsp_zero_counts(unsigned int* __restrict__ counts) {
    counts[threadIdx.x] = 0u;
}

// ---------------------------------------------------------------------------
// Kernel 1: region ids (int64 output), u8 region-id sidecar for the GEMM
// kernel, and exact integer per-region counts via LDS histogram.
// ---------------------------------------------------------------------------
__global__ void vsp_ids(const float* __restrict__ coords,
                        long long* __restrict__ ids_out,
                        unsigned char* __restrict__ rids,
                        unsigned int* __restrict__ counts) {
    __shared__ unsigned int hist[R_DIM];
    const int t = threadIdx.x;
    hist[t] = 0u;
    __syncthreads();

    const int n = blockIdx.x * 256 + t;
    const float x = coords[2 * n + 0];
    const float y = coords[2 * n + 1];
    int xs = (int)(x * (float)SIDE);
    int ys = (int)(y * (float)SIDE);
    xs = xs < 0 ? 0 : (xs > SIDE - 1 ? SIDE - 1 : xs);
    ys = ys < 0 ? 0 : (ys > SIDE - 1 ? SIDE - 1 : ys);
    int rid = ys * SIDE + xs;
    if (rid > R_DIM - 1) rid = R_DIM - 1;

    ids_out[n] = (long long)rid;
    rids[n]    = (unsigned char)rid;
    atomicAdd(&hist[rid], 1u);
    __syncthreads();
    atomicAdd(&counts[t], hist[t]);   // integer adds: deterministic
}

// ---------------------------------------------------------------------------
// Kernel 2: segment-sum as WMMA GEMM against one-hot tiles.
//   Each wave owns (b_pair in 0..3 -> two 16-row b-tiles, r_quad in 0..3 ->
//   four 16-col r-tiles, slice in 0..127) and accumulates eight 16x16 f32
//   tiles over its 1024-point N-slice using v_wmma_f32_16x16x32_f16 with an
//   f16 hi/lo split of the f32 entropy (fp32-class accuracy; one-hot exact).
//   One-hot B tiles are built once per step and reused across both b-tiles.
//   The slice's region ids are staged global->LDS with the CDNA5 async path.
//   Partial tiles go to exclusive per-slice scratch (deterministic).
// ---------------------------------------------------------------------------
__global__ void __launch_bounds__(256)
vsp_gemm(const float* __restrict__ E,
         const unsigned char* __restrict__ rids,
         float* __restrict__ partial) {
    __shared__ unsigned char srids[SLICE_PTS];   // 1 KiB, sole LDS object (offset 0)

    const int tid    = threadIdx.x;
    const int wid    = blockIdx.x * 8 + (tid >> 5);
    const int lane   = tid & 31;
    const int laneHi = lane >> 4;      // 0 or 1
    const int laneLo = lane & 15;

    const int slice  = wid >> 4;       // 16 (b_pair, r_quad) combos per slice
    const int combo  = wid & 15;
    const int b_pair = combo >> 2;     // covers b-tiles {2p, 2p+1}
    const int rq     = combo & 3;      // covers r-tiles {4q .. 4q+3}

    // ---- async-stage this slice's 1024 region-id bytes into LDS (ASYNCcnt)
    {
        const unsigned char* gsrc = rids + slice * SLICE_PTS + tid * 4;
        unsigned int ldsOff = (unsigned int)(tid * 4);   // srids is at LDS offset 0
        asm volatile("global_load_async_to_lds_b32 %0, %1, off"
                     :: "v"(ldsOff), "v"(gsrc) : "memory");
        asm volatile("s_wait_asynccnt 0" ::: "memory");
    }
    __syncthreads();

    const int b0 = b_pair * 32 + laneLo;        // A row for b-tile 0
    const float* erow0 = E + (size_t)b0 * N_PTS;
    const float* erow1 = erow0 + (size_t)16 * N_PTS;   // b-tile 1 row

    v8f acc[2][4] = {};                          // [b-tile][r-tile] f32 C/D
    const int nbase = slice * SLICE_PTS;

    for (int step = 0; step < KSTEPS; ++step) {
        const int n0 = nbase + step * 32;

        // Prefetch entropy a few k-steps ahead (global_prefetch_b8).
        __builtin_prefetch((const void*)(erow0 + n0 + 256), 0, 1);
        __builtin_prefetch((const void*)(erow1 + n0 + 256), 0, 1);

        // ---- region ids for this lane's 16 K-values from LDS (2x ds_load_b64)
        const uint2 wv0 = *(const uint2*)(srids + step * 32 + laneHi * 8);
        const uint2 wv1 = *(const uint2*)(srids + step * 32 + 16 + laneHi * 8);
        const uint32_t wbytes[4] = { wv0.x, wv0.y, wv1.x, wv1.y };

        // ---- two A tiles: E[b][n0 + K], K = (j/8)*16 + laneHi*8 + (j%8)
        v16h ahi[2], alo[2];
#pragma unroll
        for (int bt = 0; bt < 2; ++bt) {
            const float* er = bt ? erow1 : erow0;
            const float4* ap = (const float4*)(er + n0 + laneHi * 8);
            float4 f0 = ap[0];   // K = laneHi*8 + 0..3
            float4 f1 = ap[1];   // K = laneHi*8 + 4..7
            float4 f2 = ap[4];   // K = 16 + laneHi*8 + 0..3
            float4 f3 = ap[5];   // K = 16 + laneHi*8 + 4..7
            float vals[16] = { f0.x, f0.y, f0.z, f0.w, f1.x, f1.y, f1.z, f1.w,
                               f2.x, f2.y, f2.z, f2.w, f3.x, f3.y, f3.z, f3.w };
#pragma unroll
            for (int j = 0; j < 16; ++j) {
                _Float16 h = (_Float16)vals[j];
                ahi[bt][j] = h;
                alo[bt][j] = (_Float16)(vals[j] - (float)h);
            }
        }

        // ---- 4 one-hot B tiles (exact in f16), reused across both b-tiles
#pragma unroll
        for (int t = 0; t < 4; ++t) {
            const unsigned int rT = (unsigned int)((rq * 4 + t) * 16 + laneLo);
            v16h bh;
#pragma unroll
            for (int j = 0; j < 16; ++j) {
                unsigned int byte = (wbytes[j >> 2] >> ((j & 3) * 8)) & 0xFFu;
                bh[j] = (byte == rT) ? (_Float16)1.0f : (_Float16)0.0f;
            }
#pragma unroll
            for (int bt = 0; bt < 2; ++bt) {
                acc[bt][t] = __builtin_amdgcn_wmma_f32_16x16x32_f16(
                    false, ahi[bt], false, bh, (short)0, acc[bt][t], false, false);
                acc[bt][t] = __builtin_amdgcn_wmma_f32_16x16x32_f16(
                    false, alo[bt], false, bh, (short)0, acc[bt][t], false, false);
            }
        }
    }

    // ---- write exclusive per-slice partial tiles (C layout: M = i + 8*laneHi)
    float* base = partial + (size_t)slice * BR;
#pragma unroll
    for (int bt = 0; bt < 2; ++bt) {
#pragma unroll
        for (int t = 0; t < 4; ++t) {
            const int r = (rq * 4 + t) * 16 + laneLo;
#pragma unroll
            for (int i = 0; i < 8; ++i) {
                const int brow = b_pair * 32 + bt * 16 + laneHi * 8 + i;
                base[brow * R_DIM + r] = acc[bt][t][i];
            }
        }
    }
}

// ---------------------------------------------------------------------------
// Kernel 3: fixed-order slice reduction + divide by exact counts.
// ---------------------------------------------------------------------------
__global__ void vsp_norm(const float* __restrict__ partial,
                         const unsigned int* __restrict__ counts,
                         float* __restrict__ wout) {
    const int t = blockIdx.x * 256 + threadIdx.x;   // 0..BR-1, t = b*R + r
    const int r = t & (R_DIM - 1);
    float s = 0.0f;
#pragma unroll 8
    for (int sl = 0; sl < S_SLICES; ++sl) s += partial[(size_t)sl * BR + t];
    const unsigned int cnt = counts[r];
    wout[t] = (cnt > 0u) ? s / (float)cnt : 0.0f;
}

// ---------------------------------------------------------------------------
// Launch
// ---------------------------------------------------------------------------
extern "C" void kernel_launch(void* const* d_in, const int* in_sizes, int n_in,
                              void* d_out, int out_size, void* d_ws, size_t ws_size,
                              hipStream_t stream) {
    (void)in_sizes; (void)n_in; (void)out_size; (void)ws_size;

    const float* coords  = (const float*)d_in[0];
    const float* entropy = (const float*)d_in[1];
    // d_in[2] = num_regions scalar (256), baked in as R_DIM.

    long long* ids_out = (long long*)d_out;
    float*     wout    = (float*)((char*)d_out + (size_t)N_PTS * sizeof(long long));

    // Workspace layout: [ partial f32[128*B*R] | counts u32[256] | rids u8[N] ]
    const size_t partialBytes = (size_t)S_SLICES * BR * sizeof(float);   // 16 MiB
    float*         partial = (float*)d_ws;
    unsigned int*  counts  = (unsigned int*)((char*)d_ws + partialBytes);
    unsigned char* rids    = (unsigned char*)((char*)d_ws + partialBytes + 1024);

    vsp_zero_counts<<<1, 256, 0, stream>>>(counts);
    vsp_ids<<<N_PTS / 256, 256, 0, stream>>>(coords, ids_out, rids, counts);
    vsp_gemm<<<(4 * 4 * S_SLICES) / 8, 256, 0, stream>>>(entropy, rids, partial);
    vsp_norm<<<BR / 256, 256, 0, stream>>>(partial, counts, wout);
}
